// GPUOptimizedLNN_50929722196026
// MI455X (gfx1250) — matrix operand order repaired and checked by reading, not compiled, not run on hardware
//
#include <hip/hip_runtime.h>
#include <math.h>

// ---------------------------------------------------------------------------
// GPUOptimizedLNN for MI455X (gfx1250, wave32, WMMA bf16 16x16x32)
//
// Pipeline:
//   1. convert weights (pw, W_rec, W_in, ow) fp32 -> bf16 in ws
//   2. complexity net -> alpha[3][64] (per-layer, per-batch leak factor)
//   3. h_seq = x @ pw^T + pb                 (WMMA gemm, A fp32 inline-cvt)
//   4. per layer: ext = h_seq @ Wi^T + b     (WMMA gemm)
//                 scan: h = a*h + (1-a)*tanh(h @ Wr^T + ext)
//                 (WMMA, Wr in regs, h-prev in regs (fp32), ping-pong LDS h,
//                  1 barrier/step, ext software-pipelined, native v_tanh_f32)
//   5. out = h_seq[:, -1] @ ow^T + ob -> d_out (fp32)
// ---------------------------------------------------------------------------

typedef __attribute__((ext_vector_type(16))) __bf16 v16bf;
typedef __attribute__((ext_vector_type(8)))  float  v8f;

#define HID   256
#define SLEN  2048
#define BATCH 64
#define INSZ  128
#define OUTSZ 64

// Branch-free tanh: native gfx1250 v_tanh_f32 when the builtin exists,
// otherwise a single v_exp_f32 + rcp chain (no divergent libm path).
static __device__ inline float fast_tanh(float x) {
#if __has_builtin(__builtin_amdgcn_tanhf)
  return __builtin_amdgcn_tanhf(x);
#elif __has_builtin(__builtin_amdgcn_tanh_f32)
  return __builtin_amdgcn_tanh_f32(x);
#else
  float e = __expf(2.0f * x);          // v_exp_f32 (fast), branch-free
  return 1.0f - 2.0f / (e + 1.0f);     // -> [-1, 1], correct at +-inf
#endif
}

// ---------------------------------------------------------------------------
// fp32 -> bf16 elementwise convert
// ---------------------------------------------------------------------------
__global__ void f32_to_bf16_kernel(const float* __restrict__ in,
                                   __bf16* __restrict__ out, int n) {
  int i = blockIdx.x * blockDim.x + threadIdx.x;
  if (i < n) out[i] = (__bf16)in[i];
}

// ---------------------------------------------------------------------------
// complexity net + per-layer alpha.  One block per batch row, 128 threads.
// ---------------------------------------------------------------------------
__global__ void complexity_alpha_kernel(const float* __restrict__ x,
                                        const float* __restrict__ cw1,
                                        const float* __restrict__ cb1,
                                        const float* __restrict__ cw2,
                                        const float* __restrict__ cb2,
                                        const float* __restrict__ tau_bands,
                                        const float* __restrict__ tmw,
                                        const float* __restrict__ tmb,
                                        float* __restrict__ alpha_out) {
  __shared__ float smean[INSZ];
  __shared__ float shid[64];
  int b = blockIdx.x;
  int t = threadIdx.x;               // 0..127, one feature channel each
  const float* xb = x + (size_t)b * SLEN * INSZ;
  float s = 0.f;
  for (int tt = 0; tt < SLEN; ++tt) s += xb[(size_t)tt * INSZ + t];
  smean[t] = s * (1.0f / (float)SLEN);
  __syncthreads();
  if (t < 64) {
    float a = cb1[t];
    for (int c = 0; c < INSZ; ++c) a += smean[c] * cw1[t * INSZ + c];
    shid[t] = fmaxf(a, 0.f);
  }
  __syncthreads();
  if (t == 0) {
    float a = cb2[0];
    for (int j = 0; j < 64; ++j) a += shid[j] * cw2[j];
    float comp = 1.f / (1.f + expf(-a));
    for (int l = 0; l < 3; ++l) {
      float lg[4], mx = -1e30f;
      for (int k = 0; k < 4; ++k) {
        lg[k] = comp * tmw[l * 4 + k] + tmb[l * 4 + k];
        mx = fmaxf(mx, lg[k]);
      }
      float den = 0.f, num = 0.f;
      for (int k = 0; k < 4; ++k) {
        float e = expf(lg[k] - mx);
        den += e;
        num += tau_bands[l * 4 + k] * e;
      }
      float tau = num / den;
      alpha_out[l * BATCH + b] = expf(-0.05f / tau);
    }
  }
}

// ---------------------------------------------------------------------------
// Generic C = A @ W^T + bias GEMM via v_wmma_f32_16x16x32_bf16.
// 256 threads = 8 waves, one 16x16 output tile per wave.
// A: [M,K] (fp32 or bf16, leading dim lda), W: [N,K] bf16 row-major (= B^T),
// Out: [M,N] (bf16 or fp32, leading dim ldo).
// Fragment layouts per CDNA5 ISA 7.12.2 (wave32).
// ---------------------------------------------------------------------------
template <bool A_F32, bool OUT_F32>
__global__ void __launch_bounds__(256)
gemm_wmma_kernel(const void* __restrict__ Av, const __bf16* __restrict__ W,
                 const float* __restrict__ bias, void* __restrict__ Outv,
                 int M, int N, int K, long lda, long ldw, long ldo) {
  int wave = threadIdx.x >> 5;
  int lane = threadIdx.x & 31;
  int tilesN = N >> 4;
  long tile = (long)blockIdx.x * (blockDim.x >> 5) + wave;
  long tilesTotal = (long)(M >> 4) * tilesN;
  if (tile >= tilesTotal) return;
  int m0 = (int)(tile / tilesN) << 4;
  int n0 = (int)(tile % tilesN) << 4;
  int col = lane & 15;       // A row / B col / C col within tile
  int half = lane >> 4;      // lane group selects K/M sub-chunk

  v8f acc = {};
  for (int k0 = 0; k0 < K; k0 += 32) {
    v16bf a;
    if (A_F32) {
      const float* A = (const float*)Av;
      const float* p0 = A + (long)(m0 + col) * lda + k0 + half * 8;
      const float* p1 = p0 + 16;
#pragma unroll
      for (int i = 0; i < 8; ++i) a[i] = (__bf16)p0[i];
#pragma unroll
      for (int i = 0; i < 8; ++i) a[8 + i] = (__bf16)p1[i];
    } else {
      const __bf16* A = (const __bf16*)Av;
      const __bf16* p0 = A + (long)(m0 + col) * lda + k0 + half * 8;
      const __bf16* p1 = p0 + 16;
#pragma unroll
      for (int i = 0; i < 8; ++i) a[i] = p0[i];
#pragma unroll
      for (int i = 0; i < 8; ++i) a[8 + i] = p1[i];
    }
    // B fragment: lane holds W[n0+col][k0 + half*8 .. +7] and [k0 + 16 + half*8 .. +7]
    v16bf bfrag;
    const __bf16* pw = W + (long)(n0 + col) * ldw + k0 + half * 8;
#pragma unroll
    for (int i = 0; i < 16; ++i) bfrag[i] = pw[i < 8 ? i : i + 8];
    acc = __builtin_amdgcn_wmma_f32_16x16x32_bf16(false, a, false, bfrag,
                                                  (short)0, acc, false, false);
  }
  float bv = bias ? bias[n0 + col] : 0.f;
#pragma unroll
  for (int r = 0; r < 8; ++r) {
    int m = m0 + r + 8 * half;           // C/D layout: VGPR r -> row r + 8*half
    long idx = (long)m * ldo + n0 + col;
    float v = acc[r] + bv;
    if (OUT_F32) ((float*)Outv)[idx] = v;
    else         ((__bf16*)Outv)[idx] = (__bf16)v;
  }
}

// ---------------------------------------------------------------------------
// Recurrent scan for one layer.
// Grid = 4 blocks (16 batch rows each), block = 512 threads = 16 waves.
// Wave w owns output columns [16w, 16w+16):
//   - Wr B-fragments (64 VGPRs) live in registers all 2048 steps
//   - the wave's own h tile is carried in fp32 registers (hprev[8]); LDS bf16
//     is only the operand feed for the other waves' A-fragments
//   - h[16][256] bf16 ping-pongs between two LDS buffers -> 1 barrier/step
//   - ext loads are software-pipelined one step ahead
//   h_new = alpha*h + (1-alpha)*tanh(h @ Wr^T + ext_t)
// ---------------------------------------------------------------------------
__global__ void __launch_bounds__(512)
scan_kernel(const __bf16* __restrict__ ext, __bf16* __restrict__ hout,
            const __bf16* __restrict__ Wr, const float* __restrict__ alpha,
            int zero_init) {
  __shared__ __bf16 hlds[2][16][HID + 8];
  int b0 = blockIdx.x * 16;
  int tid = threadIdx.x;
  int wave = tid >> 5, lane = tid & 31;
  int n0 = wave << 4;
  int col = lane & 15, half = lane >> 4;

  // ---- init hidden state (zeros for layer 0, else prev layer's last step)
  for (int idx = tid; idx < 16 * HID; idx += 512) {
    int m = idx >> 8, n = idx & (HID - 1);
    float hv = 0.f;
    if (!zero_init)
      hv = (float)hout[((long)(b0 + m) * SLEN + (SLEN - 1)) * HID + n];
    hlds[0][m][n] = (__bf16)hv;
  }

  // ---- preload all 8 Wr B-fragments for this wave's column tile (64 VGPRs)
  v16bf Breg[8];
#pragma unroll
  for (int kk = 0; kk < 8; ++kk) {
    const __bf16* p = Wr + (long)(n0 + col) * HID + kk * 32 + half * 8;
#pragma unroll
    for (int i = 0; i < 16; ++i) Breg[kk][i] = p[i < 8 ? i : i + 8];
  }

  // ---- per-output-row constants + register-resident h_prev
  float al[8], hprev[8];
  long eoff[8];
#pragma unroll
  for (int r = 0; r < 8; ++r) {
    int m = r + 8 * half;
    al[r] = alpha[b0 + m];
    eoff[r] = (long)(b0 + m) * SLEN * HID + n0 + col;
    hprev[r] = zero_init ? 0.f : (float)hout[eoff[r] + (long)(SLEN - 1) * HID];
  }

  // ---- prime the ext pipeline with step 0
  __bf16 ecur[8];
#pragma unroll
  for (int r = 0; r < 8; ++r) ecur[r] = ext[eoff[r]];

  __syncthreads();

  int p = 0;
  for (int t = 0; t < SLEN; ++t) {
    // ---- prefetch next step's ext slice (overlaps WMMA + barrier below);
    //      branchless clamp keeps the last iteration in-bounds
    long tb1 = (long)(t + 1 < SLEN ? t + 1 : t) * HID;
    __bf16 enext[8];
#pragma unroll
    for (int r = 0; r < 8; ++r) enext[r] = ext[eoff[r] + tb1];

    // ---- load all 8 A-fragments (16 ds_load_b128 pipeline ahead of
    //      8 back-to-back WMMAs -- pure D->C accumulate chain)
    v16bf afr[8];
#pragma unroll
    for (int kk = 0; kk < 8; ++kk) {
      const __bf16* p0 = &hlds[p][col][kk * 32 + half * 8];
      const __bf16* p1 = p0 + 16;
#pragma unroll
      for (int i = 0; i < 8; ++i) afr[kk][i] = p0[i];
#pragma unroll
      for (int i = 0; i < 8; ++i) afr[kk][8 + i] = p1[i];
    }
    v8f acc = {};
#pragma unroll
    for (int kk = 0; kk < 8; ++kk)
      acc = __builtin_amdgcn_wmma_f32_16x16x32_bf16(false, afr[kk], false,
                                                    Breg[kk], (short)0, acc,
                                                    false, false);

    // ---- activation + leaky blend in fp32 registers; publish bf16 to the
    //      OTHER LDS buffer (next step's A-operand) and to global h_seq
    long tb = (long)t * HID;
#pragma unroll
    for (int r = 0; r < 8; ++r) {
      int m = r + 8 * half;
      float act = fast_tanh(acc[r] + (float)ecur[r]);
      float hnew = al[r] * hprev[r] + (1.0f - al[r]) * act;
      hprev[r] = hnew;
      __bf16 hb = (__bf16)hnew;
      hlds[p ^ 1][m][n0 + col] = hb;
      hout[eoff[r] + tb] = hb;
    }
    __syncthreads();   // writes to buf p^1 visible before next step reads it
    p ^= 1;
#pragma unroll
    for (int r = 0; r < 8; ++r) ecur[r] = enext[r];
  }
}

// ---------------------------------------------------------------------------
// Host launcher
// ---------------------------------------------------------------------------
extern "C" void kernel_launch(void* const* d_in, const int* in_sizes, int n_in,
                              void* d_out, int out_size, void* d_ws, size_t ws_size,
                              hipStream_t stream) {
  const float* x         = (const float*)d_in[0];   // [64,2048,128]
  const float* cw1       = (const float*)d_in[1];   // [64,128]
  const float* cb1       = (const float*)d_in[2];   // [64]
  const float* cw2       = (const float*)d_in[3];   // [1,64]
  const float* cb2       = (const float*)d_in[4];   // [1]
  const float* pw        = (const float*)d_in[5];   // [256,128]
  const float* pb        = (const float*)d_in[6];   // [256]
  const float* W_rec     = (const float*)d_in[7];   // [3,256,256]
  const float* W_in      = (const float*)d_in[8];   // [3,256,256]
  const float* bias      = (const float*)d_in[9];   // [3,256]
  const float* tau_bands = (const float*)d_in[10];  // [3,4]
  const float* tmw       = (const float*)d_in[11];  // [3,4]
  const float* tmb       = (const float*)d_in[12];  // [3,4]
  const float* ow        = (const float*)d_in[13];  // [64,256]
  const float* ob        = (const float*)d_in[14];  // [64]

  char* ws = (char*)d_ws;
  const size_t ACT = (size_t)BATCH * SLEN * HID * 2;     // 64 MB bf16
  __bf16* buf0    = (__bf16*)(ws);                        // h_seq
  __bf16* buf1    = (__bf16*)(ws + ACT);                  // ext
  __bf16* pw_bf   = (__bf16*)(ws + 2 * ACT);
  __bf16* wrec_bf = (__bf16*)(ws + 2 * ACT + 65536);
  __bf16* win_bf  = (__bf16*)(ws + 2 * ACT + 65536 + 393216);
  __bf16* ow_bf   = (__bf16*)(ws + 2 * ACT + 65536 + 2 * 393216);
  float*  alpha_ws = (float*)(ws + 2 * ACT + 65536 + 2 * 393216 + 32768);

  // 1) weights -> bf16
  f32_to_bf16_kernel<<<(HID * INSZ + 255) / 256, 256, 0, stream>>>(pw, pw_bf, HID * INSZ);
  f32_to_bf16_kernel<<<(3 * HID * HID + 255) / 256, 256, 0, stream>>>(W_rec, wrec_bf, 3 * HID * HID);
  f32_to_bf16_kernel<<<(3 * HID * HID + 255) / 256, 256, 0, stream>>>(W_in, win_bf, 3 * HID * HID);
  f32_to_bf16_kernel<<<(OUTSZ * HID + 255) / 256, 256, 0, stream>>>(ow, ow_bf, OUTSZ * HID);

  // 2) complexity net -> alpha[3][64]
  complexity_alpha_kernel<<<BATCH, 128, 0, stream>>>(x, cw1, cb1, cw2, cb2,
                                                     tau_bands, tmw, tmb, alpha_ws);

  // 3) h_seq0 = x @ pw^T + pb   (M = 131072, N = 256, K = 128)
  {
    const int M = BATCH * SLEN;
    long tiles = (long)(M / 16) * (HID / 16);
    int blocks = (int)((tiles + 7) / 8);
    gemm_wmma_kernel<true, false><<<blocks, 256, 0, stream>>>(
        x, pw_bf, pb, buf0, M, HID, INSZ, INSZ, INSZ, HID);
  }

  // 4) layers
  for (int l = 0; l < 3; ++l) {
    const int M = BATCH * SLEN;
    long tiles = (long)(M / 16) * (HID / 16);
    int blocks = (int)((tiles + 7) / 8);
    // ext = h_seq @ Wi^T + b
    gemm_wmma_kernel<false, false><<<blocks, 256, 0, stream>>>(
        buf0, win_bf + (size_t)l * HID * HID, bias + l * HID, buf1,
        M, HID, HID, HID, HID, HID);
    // recurrent scan (4 WGs x 16 batch rows, 16 waves each)
    scan_kernel<<<4, 512, 0, stream>>>(buf1, buf0,
                                       wrec_bf + (size_t)l * HID * HID,
                                       alpha_ws + l * BATCH, (l == 0) ? 1 : 0);
  }

  // 5) out = h_seq[:, -1] @ ow^T + ob  (M = 64, N = 64, K = 256, fp32 out)
  {
    long tiles = (BATCH / 16) * (OUTSZ / 16);      // 16 tiles
    int blocks = (int)((tiles + 7) / 8);           // 2 blocks
    gemm_wmma_kernel<false, true><<<blocks, 256, 0, stream>>>(
        buf0 + (long)(SLEN - 1) * HID, ow_bf, ob, d_out,
        BATCH, OUTSZ, HID, (long)SLEN * HID, HID, OUTSZ);
  }
}
